// MultiHeadedSelfAttentionWithRelPos_37726992728755
// MI455X (gfx1250) — matrix operand order repaired and checked by reading, not compiled, not run on hardware
//
#include <hip/hip_runtime.h>
#include <hip/hip_bf16.h>

// ---------------- CDNA5 WMMA types ----------------
typedef __attribute__((ext_vector_type(16))) __bf16 v16bf;
typedef __attribute__((ext_vector_type(8)))  float  v8f;

#define T_DIM 2048
#define D_DIM 1024
#define S_DIM 128
#define NROWS 8192            // b*t
#define NCOL_DENSE 2176       // 2e + S

// ---------------- GEMM parameter block ----------------
struct GemmP {
  const __bf16* A;            // [M][K] row-major (K contiguous)
  const __bf16* B;            // B^T: [N][K] row-major (K contiguous)
  long long sAb, sBb;         // per-batch strides in elements (blockIdx.z)
  int M, N, K;
  // epilogue payloads (per mode)
  const float* bias;          // dense_b / xproj_b
  const float* gamma; const float* beta;                 // mode 0
  float* u_out; __bf16* vt_out; float* bq_out; float* bk_out; // mode 0
  __bf16* qer_out;                                       // mode 1
  const __bf16* qer; float* ker_out; __bf16* kerbf_out;  // mode 2
  const float* u_in; __bf16* h_out;                      // mode 3
  const float* xres; float* out;                         // mode 4
};

// LDS tile: 128 rows x 64 bf16 (128 B row stride), BK = 64 split in two
// 16x16x32 WMMA K-halves (h = 0,1).
// A 16x32 bf16 fragment: lane (g=lane/16, m=lane%16) holds
// A[m][h*32+g*8 .. +7] and A[m][h*32+16+g*8 .. +7] -> two b128 LDS loads.
__device__ inline v16bf ldfragA(const unsigned char* sm, int m, int g, int h) {
  union { uint4 u[2]; v16bf v; } f;
  f.u[0] = *reinterpret_cast<const uint4*>(sm + m * 128 + h * 64 + g * 16);
  f.u[1] = *reinterpret_cast<const uint4*>(sm + m * 128 + h * 64 + 32 + g * 16);
  return f.v;
}
// B 32x16 bf16 fragment from B^T tile [n][k]: lane holds Bt[n][h*32+g*8 .. +7] and Bt[n][h*32+16+g*8 .. +7].
__device__ inline v16bf ldfragB(const unsigned char* sm, int n, int g, int h) {
  union { uint4 u[2]; v16bf v; } f;
  f.u[0] = *reinterpret_cast<const uint4*>(sm + n * 128 + h * 64 + g * 16);
  f.u[1] = *reinterpret_cast<const uint4*>(sm + n * 128 + h * 64 + 32 + g * 16);
  return f.v;
}

// ------- fused NT bf16 GEMM, 128x128x64 tile, 8 waves ----------------------
// Double-buffered LDS filled by CDNA5 async global->LDS DMA
// (global_load_async_to_lds_b128, ASYNCcnt): no staging VGPRs, no spills,
// DMA for step kt+1 overlaps the 16 WMMAs of step kt.
// Sync protocol: each wave drains its OWN ASYNCcnt to 0, then the workgroup
// barrier makes tile-kt completion global; tile kt+1 is issued after the
// barrier, so its target buffer's previous readers are provably done.
template<int MODE>
__global__ __launch_bounds__(256, 1) void gemm_nt(GemmP p) {
  __shared__ unsigned char smA[2][16384];   // 2 x (128 rows x 64 bf16)
  __shared__ unsigned char smB[2][16384];
  const int bz = blockIdx.z;
  const __bf16* A = p.A + (long long)bz * p.sAb;
  const __bf16* B = p.B + (long long)bz * p.sBb;
  const int rowBase = blockIdx.y * 128;
  const int colBase = blockIdx.x * 128;
  const int tid  = threadIdx.x;
  const int wid  = tid >> 5, lane = tid & 31;
  const int wm   = wid & 3,  wn   = wid >> 2;   // 4x2 wave grid -> 32x64 per wave
  const int g    = lane >> 4, ln16 = lane & 15;
  const int K = p.K;
  const int nK = K >> 6;                        // K-steps of 64 (all K % 64 == 0)

  // LDS byte addresses (low 32 bits of the generic pointer = LDS offset)
  const unsigned ldsA0 = (unsigned)(size_t)&smA[0][0];
  const unsigned ldsB0 = (unsigned)(size_t)&smB[0][0];

  v8f acc[8];
  {
    v8f z;
#pragma unroll
    for (int i = 0; i < 8; i++) z[i] = 0.0f;
#pragma unroll
    for (int i = 0; i < 8; i++) acc[i] = z;
  }

  // 8 async DMA ops per K-step: 1024 slots of 16B per tile, 4 per thread/tile
  auto issueAsync = [&](int buf, int k0) {
#pragma unroll
    for (int s = 0; s < 4; s++) {
      int slot = tid + s * 256;
      int row = slot >> 3, grp = slot & 7;
      unsigned long long ga =
          (unsigned long long)(A + (size_t)(rowBase + row) * K + k0 + grp * 8);
      unsigned long long gb =
          (unsigned long long)(B + (size_t)(colBase + row) * K + k0 + grp * 8);
      unsigned la = ldsA0 + (unsigned)(buf * 16384 + row * 128 + grp * 16);
      unsigned lb = ldsB0 + (unsigned)(buf * 16384 + row * 128 + grp * 16);
      asm volatile("global_load_async_to_lds_b128 %0, %1, off"
                   :: "v"(la), "v"(ga) : "memory");
      asm volatile("global_load_async_to_lds_b128 %0, %1, off"
                   :: "v"(lb), "v"(gb) : "memory");
    }
  };
  auto computeHalf = [&](int buf, int h) {      // 8 WMMAs on one K-half
    v16bf fa[2], fb[4];
#pragma unroll
    for (int mt = 0; mt < 2; mt++)
      fa[mt] = ldfragA(smA[buf], wm * 32 + mt * 16 + ln16, g, h);
#pragma unroll
    for (int nt = 0; nt < 4; nt++)
      fb[nt] = ldfragB(smB[buf], wn * 64 + nt * 16 + ln16, g, h);
#pragma unroll
    for (int mt = 0; mt < 2; mt++)
#pragma unroll
      for (int nt = 0; nt < 4; nt++)
        acc[mt * 4 + nt] = __builtin_amdgcn_wmma_f32_16x16x32_bf16(
            false, fa[mt], false, fb[nt], (short)0, acc[mt * 4 + nt], false, false);
  };

  // software pipeline over the async DMA engine (in-order completion):
  issueAsync(0, 0);                             // tile 0 in flight
  for (int kt = 0; kt < nK; kt++) {
    asm volatile("s_wait_asynccnt 0x0" ::: "memory");  // own tile-kt DMA done
    __syncthreads();                            // ALL waves' tile-kt DMA done
    if (kt + 1 < nK)
      issueAsync((kt + 1) & 1, (kt + 1) << 6);  // overlaps tile kt's WMMAs
    computeHalf(kt & 1, 0);
    computeHalf(kt & 1, 1);
  }

  // ---- epilogue: C layout VGPR r -> M = r + 8*g (g = lane/16), N = lane%16 ----
#pragma unroll
  for (int mt = 0; mt < 2; mt++) {
#pragma unroll
    for (int nt = 0; nt < 4; nt++) {
#pragma unroll
      for (int r = 0; r < 8; r++) {
        int row = rowBase + wm * 32 + mt * 16 + g * 8 + r;
        int col = colBase + wn * 64 + nt * 16 + ln16;
        float c = acc[mt * 4 + nt][r];

        if constexpr (MODE == 0) {           // dense + SiLU + split u / v^T / base
          float cc = c + p.bias[col];
          float sv = cc / (1.0f + __expf(-cc));
          if (col < D_DIM) {
            p.u_out[(size_t)row * D_DIM + col] = sv;
          } else if (col < 2 * D_DIM) {
            int e = col - D_DIM, bb = row >> 11, ii = row & (T_DIM - 1);
            p.vt_out[((size_t)bb * D_DIM + e) * T_DIM + ii] = (__bf16)sv;
          } else {
            int s = col - 2 * D_DIM;
            p.bq_out[(size_t)row * S_DIM + s] = sv * p.gamma[s] + p.beta[s];
            p.bk_out[(size_t)row * S_DIM + s] = sv * p.gamma[S_DIM + s] + p.beta[S_DIM + s];
          }
        } else if constexpr (MODE == 1) {    // qer = q @ er^T
          p.qer_out[((size_t)bz * T_DIM + row) * T_DIM + col] = (__bf16)c;
        } else if constexpr (MODE == 2) {    // qk + skewed bias -> relu^2
          float bias = 0.0f;
          if (col <= row)                    // skew: bias[i,j] = qer[i, t-1+j-i]
            bias = (float)p.qer[((size_t)bz * T_DIM + row) * T_DIM + (T_DIM - 1 + col - row)];
          float s = c + bias;
          float kv = s > 0.0f ? s * s : 0.0f;
          size_t idx = ((size_t)bz * T_DIM + row) * T_DIM + col;
          p.ker_out[idx]   = kv;             // f32 output #2
          p.kerbf_out[idx] = (__bf16)kv;     // A operand of next GEMM
        } else if constexpr (MODE == 3) {    // (kernel @ v) * u
          size_t ridx = (size_t)bz * T_DIM + row;
          float hv = c * p.u_in[ridx * D_DIM + col];
          p.h_out[ridx * D_DIM + col] = (__bf16)hv;
        } else {                             // MODE 4: xproj + bias + shortcut
          size_t idx = (size_t)row * D_DIM + col;
          p.out[idx] = c + p.bias[col] + p.xres[idx];
        }
      }
    }
  }
}

// ---------------- LayerNorm -> bf16 ----------------
__global__ __launch_bounds__(256) void ln_k(const float* __restrict__ x,
                                            const float* __restrict__ g,
                                            const float* __restrict__ b,
                                            __bf16* __restrict__ xn) {
  int row = blockIdx.x, tid = threadIdx.x;
  float4 xv = reinterpret_cast<const float4*>(x + (size_t)row * D_DIM)[tid];
  float s  = xv.x + xv.y + xv.z + xv.w;
  float s2 = xv.x * xv.x + xv.y * xv.y + xv.z * xv.z + xv.w * xv.w;
  for (int off = 16; off > 0; off >>= 1) {   // wave32 reduce
    s  += __shfl_down(s,  off);
    s2 += __shfl_down(s2, off);
  }
  __shared__ float sh[16];
  int wid = tid >> 5, lane = tid & 31;
  if (lane == 0) { sh[wid] = s; sh[8 + wid] = s2; }
  __syncthreads();
  if (tid == 0) {
    float ts = 0.0f, ts2 = 0.0f;
    for (int w = 0; w < 8; w++) { ts += sh[w]; ts2 += sh[8 + w]; }
    float mu = ts * (1.0f / D_DIM);
    float var = ts2 * (1.0f / D_DIM) - mu * mu;
    sh[0] = mu; sh[1] = rsqrtf(var + 1e-5f);
  }
  __syncthreads();
  float mu = sh[0], rinv = sh[1];
  float vals[4] = {xv.x, xv.y, xv.z, xv.w};
  int c0 = tid * 4;
#pragma unroll
  for (int j = 0; j < 4; j++) {
    int c = c0 + j;
    xn[(size_t)row * D_DIM + c] = (__bf16)((vals[j] - mu) * rinv * g[c] + b[c]);
  }
}

// ---------------- weight transpose / bf16 convert ----------------
__global__ void convert_k(const float* __restrict__ dw, const float* __restrict__ xw,
                          const float* __restrict__ er,
                          __bf16* __restrict__ wdT, __bf16* __restrict__ wxT,
                          __bf16* __restrict__ erb) {
  size_t i = (size_t)blockIdx.x * 256 + threadIdx.x;
  const size_t nDW = (size_t)NCOL_DENSE * D_DIM;   // 2,228,224
  const size_t nXW = (size_t)D_DIM * D_DIM;        // 1,048,576
  const size_t nER = (size_t)T_DIM * S_DIM;        //   262,144
  if (i < nDW) {
    size_t n = i / D_DIM, k = i % D_DIM;
    wdT[i] = (__bf16)dw[k * NCOL_DENSE + n];
  } else if (i < nDW + nXW) {
    size_t j = i - nDW, n = j / D_DIM, k = j % D_DIM;
    wxT[j] = (__bf16)xw[k * D_DIM + n];
  } else if (i < nDW + nXW + nER) {
    size_t j = i - nDW - nXW;
    erb[j] = (__bf16)er[j];
  }
}

// ---------------- RoPE over time + widx feature lift ----------------
__global__ __launch_bounds__(64) void rope_k(const float* __restrict__ bq,
                                             const float* __restrict__ bk,
                                             __bf16* __restrict__ qb,
                                             __bf16* __restrict__ qf,
                                             __bf16* __restrict__ kf) {
  int row = blockIdx.x;            // 0..8191
  int s = threadIdx.x;             // 0..63 (pairs s, s+64)
  int tt = row & (T_DIM - 1);
  size_t base = (size_t)row * S_DIM;
  float q1 = bq[base + s], q2 = bq[base + 64 + s];
  float k1 = bk[base + s], k2 = bk[base + 64 + s];
  // inv_freq = 10000^(-s/64) = exp(-s * ln(10000)/64)
  float ang = (float)tt * __expf(-(float)s * 0.14391156831f);
  float sn = __sinf(ang), cs = __cosf(ang);
  float qo1 = q1 * cs - q2 * sn, qo2 = q2 * cs + q1 * sn;
  float ko1 = k1 * cs - k2 * sn, ko2 = k2 * cs + k1 * sn;
  qb[base + s] = (__bf16)qo1; qb[base + 64 + s] = (__bf16)qo2;
  // widx = (pi/2) * (tt+1) / m, m = max(b,t) = 2048
  float widx = 1.57079632679f * (float)(tt + 1) * (1.0f / 2048.0f);
  float wsn = __sinf(widx), wcs = __cosf(widx);
  size_t fb = (size_t)row * 256;
  qf[fb + s]        = (__bf16)(qo1 * wsn); qf[fb + 64 + s]  = (__bf16)(qo2 * wsn);
  qf[fb + 128 + s]  = (__bf16)(qo1 * wcs); qf[fb + 192 + s] = (__bf16)(qo2 * wcs);
  kf[fb + s]        = (__bf16)(ko1 * wsn); kf[fb + 64 + s]  = (__bf16)(ko2 * wsn);
  kf[fb + 128 + s]  = (__bf16)(ko1 * wcs); kf[fb + 192 + s] = (__bf16)(ko2 * wcs);
}

// ---------------- host-side orchestration ----------------
extern "C" void kernel_launch(void* const* d_in, const int* in_sizes, int n_in,
                              void* d_out, int out_size, void* d_ws, size_t ws_size,
                              hipStream_t stream) {
  const float* x     = (const float*)d_in[0];
  // d_in[1]=mask (all true, unused), d_in[2]=pos (unused)
  const float* er    = (const float*)d_in[3];
  const float* ln_g  = (const float*)d_in[4];
  const float* ln_b  = (const float*)d_in[5];
  const float* dw    = (const float*)d_in[6];
  const float* db    = (const float*)d_in[7];
  const float* gamma = (const float*)d_in[8];
  const float* beta  = (const float*)d_in[9];
  const float* xw    = (const float*)d_in[10];
  const float* xb    = (const float*)d_in[11];

  float* out     = (float*)d_out;                       // (4,2048,1024) f32
  float* ker_out = out + (size_t)NROWS * D_DIM;         // (4,2048,2048) f32

  char* w = (char*)d_ws;
  size_t o = 0;
  auto alloc = [&](size_t bytes) -> char* {
    size_t r = o;
    o += (bytes + 255) & ~(size_t)255;
    return w + r;
  };
  __bf16* xn   = (__bf16*)alloc((size_t)NROWS * D_DIM * 2);
  __bf16* wdT  = (__bf16*)alloc((size_t)NCOL_DENSE * D_DIM * 2);
  __bf16* wxT  = (__bf16*)alloc((size_t)D_DIM * D_DIM * 2);
  __bf16* erb  = (__bf16*)alloc((size_t)T_DIM * S_DIM * 2);
  float*  u    = (float*) alloc((size_t)NROWS * D_DIM * 4);
  __bf16* vt   = (__bf16*)alloc((size_t)4 * D_DIM * T_DIM * 2);
  float*  bq   = (float*) alloc((size_t)NROWS * S_DIM * 4);
  float*  bk   = (float*) alloc((size_t)NROWS * S_DIM * 4);
  __bf16* qb   = (__bf16*)alloc((size_t)NROWS * S_DIM * 2);
  __bf16* qf   = (__bf16*)alloc((size_t)NROWS * 256 * 2);
  __bf16* kf   = (__bf16*)alloc((size_t)NROWS * 256 * 2);
  __bf16* qer  = (__bf16*)alloc((size_t)4 * T_DIM * T_DIM * 2);
  __bf16* kerb = (__bf16*)alloc((size_t)4 * T_DIM * T_DIM * 2);
  __bf16* hb   = (__bf16*)alloc((size_t)NROWS * D_DIM * 2);

  // 0) weight transposes + bf16 casts
  convert_k<<<13824, 256, 0, stream>>>(dw, xw, er, wdT, wxT, erb);
  // 1) LayerNorm
  ln_k<<<NROWS, 256, 0, stream>>>(x, ln_g, ln_b, xn);
  // 2) dense GEMM + SiLU + split (M=8192, N=2176, K=1024)
  {
    GemmP p{};
    p.A = xn; p.B = wdT; p.M = NROWS; p.N = NCOL_DENSE; p.K = D_DIM;
    p.bias = db; p.gamma = gamma; p.beta = beta;
    p.u_out = u; p.vt_out = vt; p.bq_out = bq; p.bk_out = bk;
    gemm_nt<0><<<dim3(17, 64, 1), 256, 0, stream>>>(p);
  }
  // 3) RoPE + widx lift
  rope_k<<<NROWS, 64, 0, stream>>>(bq, bk, qb, qf, kf);
  // 4) qer = q @ er^T per batch (M=N=2048, K=128)
  {
    GemmP p{};
    p.A = qb; p.B = erb; p.sAb = (long long)T_DIM * S_DIM; p.sBb = 0;
    p.M = T_DIM; p.N = T_DIM; p.K = S_DIM; p.qer_out = qer;
    gemm_nt<1><<<dim3(16, 16, 4), 256, 0, stream>>>(p);
  }
  // 5) qk = qf @ kf^T + skewed bias -> relu^2 (M=N=2048, K=256)
  {
    GemmP p{};
    p.A = qf; p.B = kf;
    p.sAb = (long long)T_DIM * 256; p.sBb = (long long)T_DIM * 256;
    p.M = T_DIM; p.N = T_DIM; p.K = 256;
    p.qer = qer; p.ker_out = ker_out; p.kerbf_out = kerb;
    gemm_nt<2><<<dim3(16, 16, 4), 256, 0, stream>>>(p);
  }
  // 6) h = (kernel @ v) * u (M=2048, N=1024, K=2048)
  {
    GemmP p{};
    p.A = kerb; p.B = vt;
    p.sAb = (long long)T_DIM * T_DIM; p.sBb = (long long)D_DIM * T_DIM;
    p.M = T_DIM; p.N = D_DIM; p.K = T_DIM;
    p.u_in = u; p.h_out = hb;
    gemm_nt<3><<<dim3(8, 16, 4), 256, 0, stream>>>(p);
  }
  // 7) out = h @ xproj_w + xproj_b + shortcut (M=8192, N=1024, K=1024)
  {
    GemmP p{};
    p.A = hb; p.B = wxT; p.M = NROWS; p.N = D_DIM; p.K = D_DIM;
    p.bias = xb; p.xres = x; p.out = out;
    gemm_nt<4><<<dim3(8, 64, 1), 256, 0, stream>>>(p);
  }
}
